// HistDRDoubleConv_84593675862236
// MI455X (gfx1250) — matrix operand order repaired and checked by reading, not compiled, not run on hardware
//
#include <hip/hip_runtime.h>

// ---------------------------------------------------------------------------
// HistDRConv double block for MI455X (gfx1250, wave32, WMMA).
// Strategy: HBM-bound problem (~150MB moved => ~6us at 23.3TB/s), so:
//  * histmap argmax collapsed once to a u8 region map (8MB -> 256KB reads)
//  * activations carried as bf16, conv done with v_wmma_f32_16x16x32_bf16
//  * per-sample/per-region 3x3 kernels pre-packed into the WMMA A-fragment
//    friendly layout [b][r][cout][Kpad] (K = ci*9 + kh*3+kw, zero padded)
//  * hard region selection = per-lane mask on f32 accumulators + wave-uniform
//    ballot skip of regions absent from a 16-pixel tile
// ---------------------------------------------------------------------------

typedef __attribute__((ext_vector_type(16))) __bf16 v16bf;
typedef __attribute__((ext_vector_type(8)))  float  v8f;

constexpr int  BN   = 4;
constexpr int  CIN  = 16;
constexpr int  CMID = 32;
constexpr int  COUT = 32;
constexpr int  Hh   = 256;
constexpr int  Wc   = 256;
constexpr int  RNUM = 8;
constexpr long HW   = (long)Hh * Wc;

constexpr size_t align256(size_t x) { return (x + 255) & ~(size_t)255; }
constexpr size_t OFF_XBF  = 0;                                               // x as bf16
constexpr size_t OFF_YBF  = align256(OFF_XBF  + (size_t)BN * CIN  * HW * 2); // bn(y1) as bf16
constexpr size_t OFF_REG  = align256(OFF_YBF  + (size_t)BN * CMID * HW * 2); // u8 region map
constexpr size_t OFF_POOL = align256(OFF_REG  + (size_t)BN * HW);            // pooled f32
constexpr size_t OFF_A    = align256(OFF_POOL + (size_t)BN * CMID * 9 * 4);  // A-frag weights
constexpr size_t OFF_Y    = align256(OFF_A    + (size_t)BN * RNUM * COUT * 288 * 2); // raw conv f32
constexpr size_t OFF_SS   = align256(OFF_Y    + (size_t)BN * CMID * HW * 4); // scale/shift
constexpr size_t WS_NEED  = align256(OFF_SS   + 2 * 64 * 4);

// ---------------------------------------------------------------- pack f32->bf16
__global__ void k_pack(const float* __restrict__ x, __bf16* __restrict__ out, long n) {
  for (long i = (long)blockIdx.x * blockDim.x + threadIdx.x; i < n;
       i += (long)gridDim.x * blockDim.x)
    out[i] = (__bf16)x[i];
}

// ---------------------------------------------------------------- region argmax
__global__ void k_regmap(const float* __restrict__ hist, unsigned char* __restrict__ regmap) {
  const long n = (long)BN * HW;
  for (long idx = (long)blockIdx.x * blockDim.x + threadIdx.x; idx < n;
       idx += (long)gridDim.x * blockDim.x) {
    const long b = idx / HW, px = idx - b * HW;
    const float* p = hist + b * RNUM * HW + px;
    int best = 0; float bv = p[0];
    #pragma unroll
    for (int r = 1; r < RNUM; ++r) {
      const float v = p[(long)r * HW];
      if (v > bv) { bv = v; best = r; }   // strict > keeps first max (argmax semantics)
    }
    regmap[idx] = (unsigned char)best;
  }
}

// ---------------------------------------------------------------- adaptive pool 3x3
__global__ void k_pool(const __bf16* __restrict__ xbf, float* __restrict__ pooled, int C) {
  int idx = blockIdx.x;
  const int j = idx % 3; idx /= 3;
  const int i = idx % 3; idx /= 3;
  const int c = idx % C;
  const int b = idx / C;
  const int h0 = (i * Hh) / 3, h1 = ((i + 1) * Hh + 2) / 3;
  const int w0 = (j * Wc) / 3, w1 = ((j + 1) * Wc + 2) / 3;
  const int nw = w1 - w0, n = (h1 - h0) * nw;
  const __bf16* p = xbf + ((long)b * C + c) * HW;
  float s = 0.f;
  for (int t = threadIdx.x; t < n; t += blockDim.x) {
    const int hh = h0 + t / nw, ww = w0 + (t - (t / nw) * nw);
    s += (float)p[(long)hh * Wc + ww];
  }
  __shared__ float red[256];
  red[threadIdx.x] = s;
  __syncthreads();
  for (int off = 128; off; off >>= 1) {
    if ((int)threadIdx.x < off) red[threadIdx.x] += red[threadIdx.x + off];
    __syncthreads();
  }
  if (threadIdx.x == 0) pooled[blockIdx.x] = red[0] / (float)n;
}

// ------------------------------------------------- per-sample region kernels -> A layout
// Abuf[b][g][o][K] bf16, K = ci*9 + (kh*3+kw), zero-padded to KPAD. blockDim.x == KPAD.
template<int CI, int CO>
__global__ void k_mkkern(const float* __restrict__ pooled,
                         const float* __restrict__ w1, const float* __restrict__ b1,
                         const float* __restrict__ w2, const float* __restrict__ b2,
                         __bf16* __restrict__ Abuf) {
  constexpr int KTOT = 9 * CI;
  constexpr int KPAD = ((KTOT + 31) / 32) * 32;
  const int o = blockIdx.x % CO;
  const int g = (blockIdx.x / CO) % RNUM;
  const int b = blockIdx.x / (CO * RNUM);
  const int tid = threadIdx.x;

  __shared__ float ts[RNUM][9];
  if (tid < RNUM * 9) {
    const int rr = tid / 9, ij = tid - rr * 9;
    float acc = b1[g * RNUM + rr];
    const float* wrow = w1 + (g * RNUM + rr) * CI;
    const float* prow = pooled + (long)b * CI * 9 + ij;
    for (int c = 0; c < CI; ++c) acc += wrow[c] * prow[c * 9];
    ts[rr][ij] = 1.0f / (1.0f + __expf(-acc));   // sigmoid
  }
  __syncthreads();

  __bf16 val = (__bf16)0.0f;
  if (tid < KTOT) {
    const int ci = tid / 9, k = tid - ci * 9;
    const int m = o * CI + ci;                   // torch grouped order: (out, in)
    float acc = b2[(long)g * (CI * CO) + m];
    const float* wrow2 = w2 + ((long)g * (CI * CO) + m) * RNUM;
    #pragma unroll
    for (int rr = 0; rr < RNUM; ++rr) acc += wrow2[rr] * ts[rr][k];
    val = (__bf16)acc;
  }
  Abuf[(((long)b * RNUM + g) * CO + o) * KPAD + tid] = val;
}

// ---------------------------------------------------------------- main WMMA conv
// One wave = 16 consecutive pixels (one row) x all 32 output channels.
template<int CI, int CO>
__global__ __launch_bounds__(256)
void k_conv(const __bf16* __restrict__ xbf, const __bf16* __restrict__ Abuf,
            const unsigned char* __restrict__ regmap, float* __restrict__ yout) {
  static_assert(CO == 32, "two M=16 tiles assumed");
  constexpr int KTOT = 9 * CI;
  constexpr int NCH  = (KTOT + 31) / 32;
  constexpr int KPAD = NCH * 32;

  const int lane = threadIdx.x & 31;
  const int wv   = threadIdx.x >> 5;
  const int ntw  = Wc / 16;
  int task = blockIdx.x * 8 + wv;
  const int b = task / (Hh * ntw); task -= b * (Hh * ntw);
  const int h = task / ntw;
  const int w0 = (task - h * ntw) * 16;

  const int col   = lane & 15;          // B/D column == pixel within tile
  const int wp    = w0 + col;
  const int khalf = (lane >> 4) * 16;   // B-fragment: half-wave K split

  // Build im2col B-fragments once (registers). Boundary pixels -> 0.
  const __bf16* xb = xbf + (long)b * CI * HW;
  v16bf Bf[NCH];
  #pragma unroll
  for (int c = 0; c < NCH; ++c) {
    #pragma unroll
    for (int t = 0; t < 16; ++t) {
      const int K = c * 32 + khalf + t;
      __bf16 v = (__bf16)0.0f;
      if (K < KTOT) {
        const int ci = K / 9, k = K - 9 * ci;
        const int hh = h + k / 3 - 1;
        const int ww = wp + (k % 3) - 1;
        if (hh >= 0 && hh < Hh && ww >= 0 && ww < Wc)
          v = xb[(long)ci * HW + (long)hh * Wc + ww];
      }
      Bf[c][t] = v;
    }
  }

  const int myreg = regmap[(long)b * HW + (long)h * Wc + wp];

  // A-fragment addressing: lanes<16 hold K 0..15, lanes>=16 hold 16..31.
  const int arow = lane & 15;
  const int asub = (lane >> 4) * 16;
  const __bf16* Ab = Abuf + (long)b * RNUM * CO * KPAD;

  v8f f0 = {}; v8f f1 = {};
  #pragma unroll 1
  for (int r = 0; r < RNUM; ++r) {
    // wave-uniform skip: region not present in this tile -> no GEMM
    if (__builtin_amdgcn_ballot_w32(myreg == r) == 0u) continue;
    const __bf16* Ar = Ab + (long)r * CO * KPAD;
    v8f a0 = {}; v8f a1 = {};
    #pragma unroll
    for (int c = 0; c < NCH; ++c) {
      const __bf16* p0 = Ar + arow * KPAD        + c * 32 + asub;
      const __bf16* p1 = Ar + (16 + arow) * KPAD + c * 32 + asub;
      v16bf A0, A1;
      ((uint4*)&A0)[0] = *(const uint4*)(p0);
      ((uint4*)&A0)[1] = *(const uint4*)(p0 + 8);
      ((uint4*)&A1)[0] = *(const uint4*)(p1);
      ((uint4*)&A1)[1] = *(const uint4*)(p1 + 8);
      a0 = __builtin_amdgcn_wmma_f32_16x16x32_bf16(false, A0, false, Bf[c],
                                                   (short)0, a0, false, false);
      a1 = __builtin_amdgcn_wmma_f32_16x16x32_bf16(false, A1, false, Bf[c],
                                                   (short)0, a1, false, false);
    }
    const float msk = (myreg == r) ? 1.0f : 0.0f;  // per-lane == per-pixel select
    #pragma unroll
    for (int i = 0; i < 8; ++i) { f0[i] += msk * a0[i]; f1[i] += msk * a1[i]; }
  }

  // D layout: lanes<16 rows 0..7, lanes>=16 rows 8..15, column = lane&15.
  const int mb = (lane >> 4) * 8;
  float* yb = yout + (long)b * CO * HW + (long)h * Wc + wp;
  #pragma unroll
  for (int i = 0; i < 8; ++i) {
    yb[(long)(mb + i) * HW]      = f0[i];
    yb[(long)(16 + mb + i) * HW] = f1[i];
  }
}

// ---------------------------------------------------------------- BN train stats
__global__ void k_bnstats(const float* __restrict__ y, const float* __restrict__ gamma,
                          const float* __restrict__ beta, float* __restrict__ ss, int C) {
  const int c = blockIdx.x;
  float s = 0.f, s2 = 0.f;
  for (int b = 0; b < BN; ++b) {
    const float* p = y + ((long)b * C + c) * HW;
    for (long i = threadIdx.x; i < HW; i += blockDim.x) {
      const float v = p[i]; s += v; s2 += v * v;
    }
  }
  __shared__ float r1[512], r2[512];
  r1[threadIdx.x] = s; r2[threadIdx.x] = s2;
  __syncthreads();
  for (int off = 256; off; off >>= 1) {
    if ((int)threadIdx.x < off) {
      r1[threadIdx.x] += r1[threadIdx.x + off];
      r2[threadIdx.x] += r2[threadIdx.x + off];
    }
    __syncthreads();
  }
  if (threadIdx.x == 0) {
    const float inv = 1.0f / (float)(BN * HW);
    const float m = r1[0] * inv;
    const float var = r2[0] * inv - m * m;     // biased variance (training mode)
    const float sc = gamma[c] * rsqrtf(var + 1e-5f);
    ss[c] = sc; ss[C + c] = beta[c] - m * sc;
  }
}

__global__ void k_bnapply_bf16(const float* __restrict__ y, const float* __restrict__ ss,
                               __bf16* __restrict__ out, int C) {
  const long n = (long)BN * C * HW;
  for (long i = (long)blockIdx.x * blockDim.x + threadIdx.x; i < n;
       i += (long)gridDim.x * blockDim.x) {
    const int c = (int)((i / HW) % C);
    float v = y[i] * ss[c] + ss[C + c];
    out[i] = (__bf16)(v > 0.f ? v : 0.f);
  }
}

__global__ void k_bnapply_f32(const float* __restrict__ y, const float* __restrict__ ss,
                              float* __restrict__ out, int C) {
  const long n = (long)BN * C * HW;
  for (long i = (long)blockIdx.x * blockDim.x + threadIdx.x; i < n;
       i += (long)gridDim.x * blockDim.x) {
    const int c = (int)((i / HW) % C);
    const float v = y[i] * ss[c] + ss[C + c];
    out[i] = v > 0.f ? v : 0.f;
  }
}

// ---------------------------------------------------------------- launcher
extern "C" void kernel_launch(void* const* d_in, const int* in_sizes, int n_in,
                              void* d_out, int out_size, void* d_ws, size_t ws_size,
                              hipStream_t stream) {
  const float* x    = (const float*)d_in[0];
  const float* hist = (const float*)d_in[1];
  const float* w1_1 = (const float*)d_in[2];
  const float* b1_1 = (const float*)d_in[3];
  const float* w2_1 = (const float*)d_in[4];
  const float* b2_1 = (const float*)d_in[5];
  const float* g1   = (const float*)d_in[6];
  const float* be1  = (const float*)d_in[7];
  const float* w1_2 = (const float*)d_in[8];
  const float* b1_2 = (const float*)d_in[9];
  const float* w2_2 = (const float*)d_in[10];
  const float* b2_2 = (const float*)d_in[11];
  const float* g2   = (const float*)d_in[12];
  const float* be2  = (const float*)d_in[13];
  float* out = (float*)d_out;

  if (ws_size < WS_NEED) return;   // ~60MB scratch required

  char* ws = (char*)d_ws;
  __bf16*        xbf    = (__bf16*)(ws + OFF_XBF);
  __bf16*        ybf    = (__bf16*)(ws + OFF_YBF);
  unsigned char* regmap = (unsigned char*)(ws + OFF_REG);
  float*         pooled = (float*)(ws + OFF_POOL);
  __bf16*        Abuf   = (__bf16*)(ws + OFF_A);
  float*         yraw   = (float*)(ws + OFF_Y);
  float*         ss     = (float*)(ws + OFF_SS);

  const int convBlocks = BN * Hh * (Wc / 16) / 8;          // 8 waves/block

  k_pack  <<<2048, 256, 0, stream>>>(x, xbf, (long)BN * CIN * HW);
  k_regmap<<<1024, 256, 0, stream>>>(hist, regmap);

  // ---- layer 1: CIN -> CMID
  k_pool<<<BN * CIN * 9, 256, 0, stream>>>(xbf, pooled, CIN);
  k_mkkern<CIN, CMID><<<BN * RNUM * CMID, ((9 * CIN + 31) / 32) * 32, 0, stream>>>(
      pooled, w1_1, b1_1, w2_1, b2_1, Abuf);
  k_conv<CIN, CMID><<<convBlocks, 256, 0, stream>>>(xbf, Abuf, regmap, yraw);
  k_bnstats<<<CMID, 512, 0, stream>>>(yraw, g1, be1, ss, CMID);
  k_bnapply_bf16<<<2048, 256, 0, stream>>>(yraw, ss, ybf, CMID);

  // ---- layer 2: CMID -> COUT
  k_pool<<<BN * CMID * 9, 256, 0, stream>>>(ybf, pooled, CMID);
  k_mkkern<CMID, COUT><<<BN * RNUM * COUT, ((9 * CMID + 31) / 32) * 32, 0, stream>>>(
      pooled, w1_2, b1_2, w2_2, b2_2, Abuf);
  k_conv<CMID, COUT><<<convBlocks, 256, 0, stream>>>(ybf, Abuf, regmap, yraw);
  k_bnstats<<<COUT, 512, 0, stream>>>(yraw, g2, be2, ss, COUT);
  k_bnapply_f32<<<2048, 256, 0, stream>>>(yraw, ss, out, COUT);
}
